// ImputeTransformerBlock_75471165325735
// MI455X (gfx1250) — matrix operand order repaired and checked by reading, not compiled, not run on hardware
//
#include <hip/hip_runtime.h>
#include <math.h>

// ---------------------------------------------------------------------------
// Transformer block for MI455X (gfx1250, wave32, WMMA).
// All GEMMs: bf16 operands, f32 accumulation via v_wmma_f32_16x16x32_bf16.
// GEMM waves compute 16x64 tiles (A-fragment reuse = 4x).
// Attention: flash, K/V chunk staged to LDS once per block via
// global_load_async_to_lds_b128 (ASYNCcnt), fragments read from LDS.
// ---------------------------------------------------------------------------

#define BB   4
#define SS   2048
#define DD   256
#define HH   8
#define DKK  32
#define DFF_ 1024
#define MROWS (BB * SS)   // 8192

typedef __attribute__((ext_vector_type(16))) __bf16 v16bf;
typedef __attribute__((ext_vector_type(8)))  float  v8f;

__device__ __forceinline__ unsigned short f2bf(float f) {
  unsigned x = __builtin_bit_cast(unsigned, f);
  return (unsigned short)((x + 0x7FFFu + ((x >> 16) & 1u)) >> 16); // RNE
}

// A fragment: 16x32 bf16 (M x K), row-major source, ld in elements.
// lane L -> M = L&15; K-base = (L<16 ? 0 : 16); VGPR j holds
// K = kb+2j,kb+2j+1 (packed pairs, contiguous; matches load_b16).
__device__ __forceinline__ v16bf load_a16(const unsigned short* p, int ld) {
  int lane = threadIdx.x & 31;
  int m  = lane & 15;
  int kb = (lane < 16) ? 0 : 16;
  const unsigned short* row = p + m * ld;
  union { unsigned u[8]; v16bf v; } f;
#pragma unroll
  for (int j = 0; j < 8; ++j) {
    f.u[j]     = *(const unsigned*)(row + kb + 2 * j);
  }
  return f.v;
}

// B fragment: 32x16 bf16 (K x N). Source rows indexed by N (i.e. W[n][k]
// row-major, ld in elements): lane L -> N = L&15; K-base = (L<16 ? 0 : 16);
// VGPR j holds K = kb+2j,kb+2j+1 (packed pairs, contiguous in memory).
__device__ __forceinline__ v16bf load_b16(const unsigned short* w, int ld) {
  int lane = threadIdx.x & 31;
  int n  = lane & 15;
  int kb = (lane < 16) ? 0 : 16;
  const unsigned short* row = w + n * ld;
  union { unsigned u[8]; v16bf v; } f;
#pragma unroll
  for (int j = 0; j < 8; ++j)
    f.u[j] = *(const unsigned*)(row + kb + 2 * j);
  return f.v;
}

#define WMMA_BF16(A, B, C) \
  __builtin_amdgcn_wmma_f32_16x16x32_bf16(false, (A), false, (B), (short)0, (C), false, false)

// Async global -> LDS copy, one b128 per lane (ISA: GLOBAL_LOAD_ASYNC_TO_LDS_B128,
// tracked by ASYNCcnt; LDS dest = low 32 bits of generic shared pointer).
__device__ __forceinline__ void async_g2l_b128(void* lds_ptr, const void* gptr) {
  unsigned l = (unsigned)(unsigned long long)lds_ptr;
  unsigned long long g = (unsigned long long)gptr;
  asm volatile("global_load_async_to_lds_b128 %0, %1, off"
               :: "v"(l), "v"(g) : "memory");
}
__device__ __forceinline__ void wait_async0() {
  asm volatile("s_wait_asynccnt 0" ::: "memory");
}

// ---------------------------------------------------------------------------
// Fold low-rank Q/K: W_eff = W_up @ W_down (256x256), b_eff = W_up@b_down+b_up
// Writes bf16 W_eff into wcat rows [0..255] (Q) / [256..511] (K).
// ---------------------------------------------------------------------------
__global__ void prep_eff_kernel(const float* __restrict__ Wqd, const float* __restrict__ bqd,
                                const float* __restrict__ Wqu, const float* __restrict__ bqu,
                                const float* __restrict__ Wkd, const float* __restrict__ bkd,
                                const float* __restrict__ Wku, const float* __restrict__ bku,
                                unsigned short* __restrict__ wcat, float* __restrict__ bcat) {
  int gid  = blockIdx.x * blockDim.x + threadIdx.x;  // 0..131071
  int which = gid >> 16;
  int rem   = gid & 65535;
  int o = rem >> 8, i = rem & 255;
  const float* Wu = which ? Wku : Wqu;
  const float* Wd = which ? Wkd : Wqd;
  const float* bd = which ? bkd : bqd;
  const float* bu = which ? bku : bqu;
  float s = 0.f;
#pragma unroll 8
  for (int t = 0; t < 64; ++t) s += Wu[o * 64 + t] * Wd[t * 256 + i];
  wcat[(which * 256 + o) * 256 + i] = f2bf(s);
  if (i == 0) {
    float bb = bu[o];
    for (int t = 0; t < 64; ++t) bb += Wu[o * 64 + t] * bd[t];
    bcat[which * 256 + o] = bb;
  }
}

__global__ void cvt_bf16_kernel(const float* __restrict__ in, unsigned short* __restrict__ out, int n) {
  int i = blockIdx.x * blockDim.x + threadIdx.x;
  if (i < n) out[i] = f2bf(in[i]);
}

__global__ void copy_f32_kernel(const float* __restrict__ in, float* __restrict__ out, int n) {
  int i = blockIdx.x * blockDim.x + threadIdx.x;
  if (i < n) out[i] = in[i];
}

// ---------------------------------------------------------------------------
// Fused QKV GEMM: xbf(8192x256) @ wcat^T(256x768) + bcat. One wave computes a
// 16x64 tile (4 accumulators, A-frag reused 4x). Scatter into head layouts:
// Q,K -> [B,H,S,DK] bf16 ; V -> [B,H,DK,S] bf16 (transposed).
// ---------------------------------------------------------------------------
__global__ __launch_bounds__(32) void gemm_qkv_kernel(
    const unsigned short* __restrict__ A, const unsigned short* __restrict__ Wc,
    const float* __restrict__ bc,
    unsigned short* __restrict__ qh, unsigned short* __restrict__ kh,
    unsigned short* __restrict__ vt) {
  const int r0 = blockIdx.x * 16;
  const int n0 = blockIdx.y * 64;
  v8f acc[4] = {{}, {}, {}, {}};
#pragma unroll
  for (int kk = 0; kk < 256; kk += 32) {
    v16bf a = load_a16(A + r0 * 256 + kk, 256);
#pragma unroll
    for (int t = 0; t < 4; ++t) {
      v16bf b = load_b16(Wc + (n0 + 16 * t) * 256 + kk, 256);
      acc[t] = WMMA_BF16(a, b, acc[t]);
    }
  }
  const int lane = threadIdx.x & 31;
  const int n = lane & 15;
  const int hi8 = (lane >> 4) << 3;
#pragma unroll
  for (int t = 0; t < 4; ++t) {
    const int o = n0 + 16 * t + n;
    const float bias = bc[o];
#pragma unroll
    for (int i = 0; i < 8; ++i) {
      int r = r0 + i + hi8;               // flat row = b*S + s
      int b = r >> 11, s = r & 2047;
      unsigned short v = f2bf(acc[t][i] + bias);
      if (o < 256) {
        int h = o >> 5, dk = o & 31;
        qh[(((b * HH + h) * SS) + s) * DKK + dk] = v;
      } else if (o < 512) {
        int o2 = o - 256, h = o2 >> 5, dk = o2 & 31;
        kh[(((b * HH + h) * SS) + s) * DKK + dk] = v;
      } else {
        int o2 = o - 512, h = o2 >> 5, dk = o2 & 31;
        vt[(((b * HH + h) * DKK) + dk) * SS + s] = v;
      }
    }
  }
}

// ---------------------------------------------------------------------------
// Flash attention. 4 waves/block, each wave owns a 16-row query tile; all
// waves share each 32-key K/V chunk, staged to LDS once per block via async
// global->LDS b128 copies (ASYNCcnt), then consumed as ds-loaded fragments.
// ---------------------------------------------------------------------------
__global__ __launch_bounds__(128) void attn_kernel(
    const unsigned short* __restrict__ qh, const unsigned short* __restrict__ kh,
    const unsigned short* __restrict__ vt, unsigned short* __restrict__ ctx) {
  __shared__ float          s_s[4][16][32];
  __shared__ unsigned short s_p[4][16][32];
  __shared__ float          s_a[4][16];
  __shared__ float          s_l[4][16];
  __shared__ unsigned short k_lds[32 * 32];  // [key][dk]
  __shared__ unsigned short v_lds[32 * 32];  // [dk][key]

  const int tid  = threadIdx.x;
  const int w    = tid >> 5;
  const int lane = tid & 31;
  const int n    = lane & 15;
  const int hi8  = (lane >> 4) << 3;
  const int bh   = blockIdx.y;                 // b*H + h
  const int q0   = blockIdx.x * 64 + w * 16;
  const float scale = 0.17677669529663687f;    // 1/sqrt(32)

  v16bf qf = load_a16(qh + (bh * SS + q0) * DKK, DKK);  // resident all loop
  v8f o0 = {}, o1 = {};
  float m_run = -1e30f, l_run = 0.f;

  const int vdk  = tid >> 2;                   // V staging: row dk
  const int vprt = (tid & 3) * 16;             // 16B piece within 64B row seg

  for (int k0 = 0; k0 < SS; k0 += 32) {
    // ---- stage K (2KB contiguous) + V (32 x 64B rows) chunk into LDS -----
    async_g2l_b128((char*)k_lds + tid * 16,
                   (const char*)(kh + (bh * SS + k0) * DKK) + tid * 16);
    async_g2l_b128((char*)v_lds + vdk * 64 + vprt,
                   (const char*)(vt + (bh * DKK + vdk) * SS + k0) + vprt);
    wait_async0();
    __syncthreads();

    // ---- scores: A=Q(16x32), B=K^T(32x16) twice (fragments from LDS) -----
    v16bf kf0 = load_b16(k_lds, 32);
    v16bf kf1 = load_b16(k_lds + 16 * 32, 32);
    v8f zc = {};
    v8f t0 = WMMA_BF16(qf, kf0, zc);
    v8f t1 = WMMA_BF16(qf, kf1, zc);
#pragma unroll
    for (int i = 0; i < 8; ++i) {
      s_s[w][i + hi8][n]      = t0[i];
      s_s[w][i + hi8][n + 16] = t1[i];
    }
    __syncthreads();
    if (lane < 16) {                           // one lane per query row
      int r = lane;
      float sv[32], mc = -1e30f;
#pragma unroll
      for (int c = 0; c < 32; ++c) { float v = s_s[w][r][c] * scale; sv[c] = v; mc = fmaxf(mc, v); }
      float mnew  = fmaxf(m_run, mc);
      float alpha = __expf(m_run - mnew);
      float ps = 0.f;
#pragma unroll
      for (int c = 0; c < 32; ++c) { float p = __expf(sv[c] - mnew); ps += p; s_p[w][r][c] = f2bf(p); }
      l_run = alpha * l_run + ps;
      m_run = mnew;
      s_a[w][r] = alpha;
    }
    __syncthreads();
#pragma unroll
    for (int i = 0; i < 8; ++i) {              // rescale running output
      float a = s_a[w][i + hi8];
      o0[i] *= a; o1[i] *= a;
    }
    // ---- attn @ V : A=P(16x32), B=V(32keys x 16dk) twice, from LDS -------
    v16bf pf  = load_a16(&s_p[w][0][0], 32);
    v16bf vf0 = load_b16(v_lds, 32);
    v16bf vf1 = load_b16(v_lds + 16 * 32, 32);
    o0 = WMMA_BF16(pf, vf0, o0);
    o1 = WMMA_BF16(pf, vf1, o1);
    __syncthreads();
  }
  if (lane < 16) s_l[w][lane] = l_run;
  __syncthreads();
  const int b = bh >> 3, h = bh & 7;
#pragma unroll
  for (int i = 0; i < 8; ++i) {
    int m = i + hi8;
    float inv = 1.f / s_l[w][m];
    int base = (b * SS + (q0 + m)) * DD + h * DKK;
    ctx[base + n]      = f2bf(o0[i] * inv);
    ctx[base + n + 16] = f2bf(o1[i] * inv);
  }
}

// ---------------------------------------------------------------------------
// Generic bf16 GEMM -> f32 out, fused bias + optional residual add.
// A: MxK bf16 row-major ; W: NxK bf16 row-major (out = A @ W^T).
// One wave computes a 16x64 tile (N must be a multiple of 64).
// ---------------------------------------------------------------------------
__global__ __launch_bounds__(32) void gemm_f32_kernel(
    const unsigned short* __restrict__ A, const unsigned short* __restrict__ W,
    const float* __restrict__ bias, const float* __restrict__ res,
    float* __restrict__ out, int K, int N) {
  const int r0 = blockIdx.x * 16;
  const int n0 = blockIdx.y * 64;
  v8f acc[4] = {{}, {}, {}, {}};
  for (int kk = 0; kk < K; kk += 32) {
    v16bf a = load_a16(A + r0 * K + kk, K);
#pragma unroll
    for (int t = 0; t < 4; ++t) {
      v16bf b = load_b16(W + (n0 + 16 * t) * K + kk, K);
      acc[t] = WMMA_BF16(a, b, acc[t]);
    }
  }
  const int lane = threadIdx.x & 31;
  const int n = lane & 15, hi8 = (lane >> 4) << 3;
#pragma unroll
  for (int t = 0; t < 4; ++t) {
    const int o = n0 + 16 * t + n;
    const float bi = bias[o];
#pragma unroll
    for (int i = 0; i < 8; ++i) {
      int r = r0 + i + hi8;
      float v = acc[t][i] + bi;
      if (res) v += res[r * N + o];
      out[r * N + o] = v;
    }
  }
}

// FFN1: GEMM + bias + exact-erf GELU, bf16 out. 16x64 tile per wave.
__global__ __launch_bounds__(32) void gemm_gelu_kernel(
    const unsigned short* __restrict__ A, const unsigned short* __restrict__ W,
    const float* __restrict__ bias, unsigned short* __restrict__ out, int K, int N) {
  const int r0 = blockIdx.x * 16;
  const int n0 = blockIdx.y * 64;
  v8f acc[4] = {{}, {}, {}, {}};
  for (int kk = 0; kk < K; kk += 32) {
    v16bf a = load_a16(A + r0 * K + kk, K);
#pragma unroll
    for (int t = 0; t < 4; ++t) {
      v16bf b = load_b16(W + (n0 + 16 * t) * K + kk, K);
      acc[t] = WMMA_BF16(a, b, acc[t]);
    }
  }
  const int lane = threadIdx.x & 31;
  const int n = lane & 15, hi8 = (lane >> 4) << 3;
#pragma unroll
  for (int t = 0; t < 4; ++t) {
    const int o = n0 + 16 * t + n;
    const float bi = bias[o];
#pragma unroll
    for (int i = 0; i < 8; ++i) {
      int r = r0 + i + hi8;
      float v = acc[t][i] + bi;
      float g = 0.5f * v * (1.f + erff(v * 0.70710678118654752f));  // exact GELU
      out[r * N + o] = f2bf(g);
    }
  }
}

// ---------------------------------------------------------------------------
// LayerNorm over D=256: one wave per row, shuffle reductions (wave32).
// Writes f32 and optionally a bf16 copy (as the next GEMM's input).
// ---------------------------------------------------------------------------
__global__ __launch_bounds__(256) void ln_kernel(
    const float* __restrict__ in, const float* __restrict__ g, const float* __restrict__ b,
    float* __restrict__ out, unsigned short* __restrict__ bfout) {
  const int row  = blockIdx.x * 8 + (threadIdx.x >> 5);
  const int lane = threadIdx.x & 31;
  const float* p = in + row * DD;
  float vals[8], s = 0.f, s2 = 0.f;
#pragma unroll
  for (int j = 0; j < 8; ++j) {
    float v = p[lane + 32 * j];
    vals[j] = v; s += v; s2 += v * v;
  }
#pragma unroll
  for (int off = 16; off > 0; off >>= 1) {
    s  += __shfl_xor(s,  off, 32);
    s2 += __shfl_xor(s2, off, 32);
  }
  float mean = s * (1.f / 256.f);
  float var  = s2 * (1.f / 256.f) - mean * mean;
  float rs   = rsqrtf(var + 1e-5f);
#pragma unroll
  for (int j = 0; j < 8; ++j) {
    int c = lane + 32 * j;
    float o = (vals[j] - mean) * rs * g[c] + b[c];
    out[row * DD + c] = o;
    if (bfout) bfout[row * DD + c] = f2bf(o);
  }
}

// ---------------------------------------------------------------------------
extern "C" void kernel_launch(void* const* d_in, const int* in_sizes, int n_in,
                              void* d_out, int out_size, void* d_ws, size_t ws_size,
                              hipStream_t stream) {
  const float* x       = (const float*)d_in[0];
  const float* Wq_down = (const float*)d_in[1];
  const float* bq_down = (const float*)d_in[2];
  const float* Wq_up   = (const float*)d_in[3];
  const float* bq_up   = (const float*)d_in[4];
  const float* Wk_down = (const float*)d_in[5];
  const float* bk_down = (const float*)d_in[6];
  const float* Wk_up   = (const float*)d_in[7];
  const float* bk_up   = (const float*)d_in[8];
  const float* Wv      = (const float*)d_in[9];
  const float* bv      = (const float*)d_in[10];
  const float* Wo      = (const float*)d_in[11];
  const float* bo      = (const float*)d_in[12];
  const float* ln_ag   = (const float*)d_in[13];
  const float* ln_ab   = (const float*)d_in[14];
  const float* W_ff1   = (const float*)d_in[15];
  const float* b_ff1   = (const float*)d_in[16];
  const float* W_ff2   = (const float*)d_in[17];
  const float* b_ff2   = (const float*)d_in[18];
  const float* ln_bg   = (const float*)d_in[19];
  const float* ln_bb   = (const float*)d_in[20];
  float* out = (float*)d_out;

  char* ws = (char*)d_ws;
  size_t off = 0;
  auto alloc = [&](size_t bytes) -> void* {
    void* p = ws + off;
    off = (off + bytes + 255) & ~(size_t)255;
    return p;
  };
  unsigned short* xbf  = (unsigned short*)alloc((size_t)MROWS * DD * 2);
  unsigned short* wcat = (unsigned short*)alloc((size_t)768 * DD * 2);
  float*          bcat = (float*)         alloc(768 * 4);
  unsigned short* qh   = (unsigned short*)alloc((size_t)MROWS * DD * 2);
  unsigned short* kh   = (unsigned short*)alloc((size_t)MROWS * DD * 2);
  unsigned short* vt   = (unsigned short*)alloc((size_t)MROWS * DD * 2);
  unsigned short* ctx  = (unsigned short*)alloc((size_t)MROWS * DD * 2);
  unsigned short* wobf = (unsigned short*)alloc((size_t)DD * DD * 2);
  unsigned short* w1bf = (unsigned short*)alloc((size_t)DFF_ * DD * 2);
  unsigned short* w2bf = (unsigned short*)alloc((size_t)DD * DFF_ * 2);
  float*          y1   = (float*)         alloc((size_t)MROWS * DD * 4);
  float*          aof  = (float*)         alloc((size_t)MROWS * DD * 4);
  unsigned short* aob  = (unsigned short*)alloc((size_t)MROWS * DD * 2);
  unsigned short* hbf  = (unsigned short*)alloc((size_t)MROWS * DFF_ * 2);
  float*          y2   = (float*)         alloc((size_t)MROWS * DD * 4);
  (void)ws_size; (void)in_sizes; (void)n_in; (void)out_size;

  // --- weight prep: fold low-rank, convert to bf16 -------------------------
  prep_eff_kernel<<<512, 256, 0, stream>>>(Wq_down, bq_down, Wq_up, bq_up,
                                           Wk_down, bk_down, Wk_up, bk_up,
                                           wcat, bcat);
  cvt_bf16_kernel<<<(DD * DD + 255) / 256, 256, 0, stream>>>(Wv, wcat + 512 * DD, DD * DD);
  copy_f32_kernel<<<1, 256, 0, stream>>>(bv, bcat + 512, DD);
  cvt_bf16_kernel<<<(MROWS * DD + 255) / 256, 256, 0, stream>>>(x, xbf, MROWS * DD);
  cvt_bf16_kernel<<<(DD * DD + 255) / 256, 256, 0, stream>>>(Wo, wobf, DD * DD);
  cvt_bf16_kernel<<<(DFF_ * DD + 255) / 256, 256, 0, stream>>>(W_ff1, w1bf, DFF_ * DD);
  cvt_bf16_kernel<<<(DD * DFF_ + 255) / 256, 256, 0, stream>>>(W_ff2, w2bf, DD * DFF_);

  // --- fused QKV GEMM + head scatter (16x64 tiles) -------------------------
  gemm_qkv_kernel<<<dim3(MROWS / 16, 768 / 64), 32, 0, stream>>>(xbf, wcat, bcat, qh, kh, vt);

  // --- flash attention (async K/V staging) ---------------------------------
  attn_kernel<<<dim3(SS / 64, BB * HH), 128, 0, stream>>>(qh, kh, vt, ctx);

  // --- output projection + residual(x) -------------------------------------
  gemm_f32_kernel<<<dim3(MROWS / 16, DD / 64), 32, 0, stream>>>(ctx, wobf, bo, x, y1, DD, DD);

  // --- LN1 (writes f32 for residual + bf16 for FFN input) ------------------
  ln_kernel<<<MROWS / 8, 256, 0, stream>>>(y1, ln_ag, ln_ab, aof, aob);

  // --- FFN: GEMM+GELU, then GEMM + residual(attn_out) ----------------------
  gemm_gelu_kernel<<<dim3(MROWS / 16, DFF_ / 64), 32, 0, stream>>>(aob, w1bf, b_ff1, hbf, DD, DFF_);
  gemm_f32_kernel<<<dim3(MROWS / 16, DD / 64), 32, 0, stream>>>(hbf, w2bf, b_ff2, aof, y2, DFF_, DD);

  // --- final LN to d_out ---------------------------------------------------
  ln_kernel<<<MROWS / 8, 256, 0, stream>>>(y2, ln_bg, ln_bb, out, nullptr);
}
